// MusicAwareQuantizer_86277303042186
// MI455X (gfx1250) — matrix operand order repaired and checked by reading, not compiled, not run on hardware
//
#include <hip/hip_runtime.h>
#include <math.h>

#define BB 16
#define DD 256
#define TT 2048
#define KK 1024
#define MM (BB*TT)   // 32768 rows of x_flat

typedef __attribute__((ext_vector_type(16))) _Float16 v16h;
typedef __attribute__((ext_vector_type(8)))  _Float16 v8h;
typedef __attribute__((ext_vector_type(8)))  float    v8f;
typedef __attribute__((ext_vector_type(4)))  unsigned int u32x4;
typedef __attribute__((ext_vector_type(8)))  int         i32x8;
typedef __attribute__((ext_vector_type(4)))  int         i32x4;

// ---- compile-time environment probe (device pass only; read from stderr) ----
#define MAQ_STR2(x) #x
#define MAQ_STR(x) MAQ_STR2(x)
#if defined(__HIP_DEVICE_COMPILE__)
#pragma message("MAQ_DIAG clang_major=" MAQ_STR(__clang_major__))
#if __has_builtin(__builtin_amdgcn_tensor_load_to_lds)
#pragma message("MAQ_DIAG has tensor_load_to_lds: YES")
#else
#pragma message("MAQ_DIAG has tensor_load_to_lds: NO")
#endif
#if __has_builtin(__builtin_amdgcn_s_wait_tensorcnt)
#pragma message("MAQ_DIAG has s_wait_tensorcnt: YES")
#else
#pragma message("MAQ_DIAG has s_wait_tensorcnt: NO")
#endif
#if __has_builtin(__builtin_amdgcn_global_load_async_to_lds_b128)
#pragma message("MAQ_DIAG has global_load_async_to_lds_b128: YES")
#else
#pragma message("MAQ_DIAG has global_load_async_to_lds_b128: NO")
#endif
#if __has_builtin(__builtin_amdgcn_s_wait_asynccnt)
#pragma message("MAQ_DIAG has s_wait_asynccnt: YES")
#else
#pragma message("MAQ_DIAG has s_wait_asynccnt: NO")
#endif
#endif

// Tensor Data Mover: builtin exists on both toolchains with different arity
// (ROCm 7.2 / clang-22: 5 args; amdgpu-toolchain / clang-23: 6 args, extra i32x8).
#if defined(__gfx1250__) && \
    __has_builtin(__builtin_amdgcn_tensor_load_to_lds) && \
    __has_builtin(__builtin_amdgcn_s_wait_tensorcnt)
#define HAS_TDM 1
#else
#define HAS_TDM 0
#endif

// ---- workspace layout (bytes) ----
#define WS_CBH    0                      // _Float16[KK*DD]  512 KB (f16 codebook)
#define WS_C2     (512*1024)             // float[KK]        4 KB   (codebook sq norms)
#define WS_COUNTS (516*1024)             // float[KK]        4 KB   (usage histogram)
#define WS_ACCUM  (520*1024)             // float[8]                (loss accumulators)
#define WS_CODES  (520*1024 + 256)       // int[MM]          128 KB
#define WS_MSIM   (WS_CODES + MM*4)      // float[MM]        128 KB
// total ~ 776 KB

// ---------------------------------------------------------------------------
// 1) prep: f16 codebook + squared norms; zero accumulators/counters
// ---------------------------------------------------------------------------
__global__ __launch_bounds__(128) void maq_prep_kernel(const float* __restrict__ codebook,
                                                       char* __restrict__ ws) {
    _Float16* cbh   = (_Float16*)(ws + WS_CBH);
    float* c2       = (float*)(ws + WS_C2);
    float* counts   = (float*)(ws + WS_COUNTS);
    float* accum    = (float*)(ws + WS_ACCUM);
    const int n = blockIdx.x;
    const int tid = threadIdx.x;

    float s = 0.f;
    for (int d = tid; d < DD; d += 128) {
        float v = codebook[(size_t)n * DD + d];
        cbh[(size_t)n * DD + d] = (_Float16)v;
        s += v * v;
    }
    __shared__ float red[128];
    red[tid] = s; __syncthreads();
    for (int off = 64; off > 0; off >>= 1) {
        if (tid < off) red[tid] += red[tid + off];
        __syncthreads();
    }
    if (tid == 0) { c2[n] = red[0]; counts[n] = 0.f; }
    if (n == 0 && tid < 8) accum[tid] = 0.f;
}

// ---------------------------------------------------------------------------
// TDM descriptor: load one 32-row x 256-half codebook tile into LDS with
// 16B padding per 512B row (row stride 528B -> bank-conflict-free ds reads).
// ---------------------------------------------------------------------------
#if HAS_TDM
__device__ __forceinline__ void tdm_load_cb_tile(const _Float16* __restrict__ gsrc,
                                                 _Float16* ldst) {
    unsigned long long ga = (unsigned long long)(size_t)gsrc;
    unsigned int la = (unsigned int)(size_t)ldst;   // low 32 bits of LDS generic ptr = LDS byte offset
    u32x4 g0;
    g0[0] = 1u;                                        // count=1, user descriptor, no gather
    g0[1] = la;                                        // lds_addr
    g0[2] = (unsigned int)(ga & 0xFFFFFFFFu);          // global_addr[31:0]
    g0[3] = (unsigned int)((ga >> 32) & 0x1FFFFFFu)    // global_addr[56:32]
          | (2u << 30);                                // type = 2 ("image")
    i32x8 g1;
    // data_size=2B (1<<16), pad_enable (1<<20), pad_interval=128 dwords (6<<22),
    // pad_amount=4 dwords (3<<25)  -> +16B after each 512B row
    g1[0] = (int)((1u << 16) | (1u << 20) | (6u << 22) | (3u << 25));
    g1[1] = (int)(256u << 16);   // tensor_dim0 = 256 elements (bits 79:48 low half)
    g1[2] = (int)(32u << 16);    // tensor_dim0 hi = 0 | tensor_dim1 = 32 (low 16)
    g1[3] = (int)(256u << 16);   // tensor_dim1 hi = 0 | tile_dim0 = 256
    g1[4] = 32;                  // tile_dim1 = 32, tile_dim2 = 0
    g1[5] = 256;                 // tensor_dim0_stride = 256 elements (low 32)
    g1[6] = 0;                   // stride hi, tensor_dim1_stride lo
    g1[7] = 0;
    i32x4 z4 = {0, 0, 0, 0};     // groups 2/3 unused (2-D tile)
#if __clang_major__ >= 23
    i32x8 z8 = {0, 0, 0, 0, 0, 0, 0, 0};
    __builtin_amdgcn_tensor_load_to_lds(g0, g1, z4, z4, z8, 0);
#else
    __builtin_amdgcn_tensor_load_to_lds(g0, g1, z4, z4, 0);
#endif
}
#endif

// ---------------------------------------------------------------------------
// 2) WMMA GEMM (x_flat @ codebook^T) + fused argmin over K=1024
//    block = 128 threads (4 waves), covers 64 rows (one b, 64 consecutive t)
//    B tiles staged once per workgroup via TDM (double-buffered), shared by
//    all 4 waves from LDS.
// ---------------------------------------------------------------------------
__global__ __launch_bounds__(128) void maq_gemm_argmin_kernel(const float* __restrict__ x,
                                                              char* __restrict__ ws,
                                                              float* __restrict__ codes_out_f) {
    const _Float16* cbh = (const _Float16*)(ws + WS_CBH);
    const float* c2     = (const float*)(ws + WS_C2);
    int* codesI         = (int*)(ws + WS_CODES);

    // A tile staged in LDS as f16: 64 rows x 256 d, row stride 264 halves
    __shared__ __align__(16) _Float16 ldsA[64 * 264];
#if HAS_TDM
    // B double buffer: 32 rows x 264 halves (TDM pads 512B rows to 528B)
    __shared__ __align__(16) _Float16 ldsB[2][32 * 264];
#endif

    const int tid = threadIdx.x;
    const int blk = blockIdx.x;          // 512 blocks = 16 b * 32 t-chunks
    const int b   = blk >> 5;
    const int t0  = (blk & 31) * 64;

    const int lane = tid & 31;
    const int wave = tid >> 5;
    const int g    = lane >> 4;          // lane half (K-half select)
    const int nlo  = lane & 15;          // N (and M) sub-index
    const int rowLocal = wave * 16 + nlo;

#if HAS_TDM
    const bool leader = (wave == 0);     // tensor ops ignore EXEC: one DMA per wave
    if (leader) {
        tdm_load_cb_tile(cbh + (size_t)0 * 32 * DD, &ldsB[0][0]);
        tdm_load_cb_tile(cbh + (size_t)1 * 32 * DD, &ldsB[1][0]);
    }
#endif

    // stage A: coalesced along t (64 consecutive floats per d)
    for (int idx = tid; idx < 64 * DD; idx += 128) {
        int r = idx & 63;
        int d = idx >> 6;
        float v = x[((size_t)b * DD + d) * TT + t0 + r];
        ldsA[r * 264 + d] = (_Float16)v;
    }
    __syncthreads();

    // Resident A row-panel: 8 K-chunks of 32 (contiguous layout matching B:
    // K = c*32 + g*16 + h)
    v16h aF[8];
#pragma unroll
    for (int c = 0; c < 8; ++c) {
        const _Float16* base = &ldsA[rowLocal * 264 + c * 32 + g * 16];
        v8h lo = *(const v8h*)(base);
        v8h hi = *(const v8h*)(base + 8);
        aF[c] = __builtin_shufflevector(lo, hi, 0,1,2,3,4,5,6,7,8,9,10,11,12,13,14,15);
    }

    float best[8];
    int   bidx[8];
#pragma unroll
    for (int r = 0; r < 8; ++r) { best[r] = 3.4e38f; bidx[r] = 0; }

    // 32 stages of 32 codebook rows each (two 16-wide N-tiles in flight)
    for (int s = 0; s < 32; ++s) {
        const int buf = s & 1;
        const int n0 = s * 32 + nlo;
        const int n1 = n0 + 16;

#if HAS_TDM
        if (leader) {
            if (s == 31) __builtin_amdgcn_s_wait_tensorcnt((short)0);
            else         __builtin_amdgcn_s_wait_tensorcnt((short)1);
        }
        __syncthreads();                 // tile s visible to all waves
        const _Float16* bbase0 = &ldsB[buf][nlo * 264 + g * 16];
        const _Float16* bbase1 = &ldsB[buf][(nlo + 16) * 264 + g * 16];
#else
        const _Float16* bbase0 = cbh + (size_t)n0 * DD + g * 16;
        const _Float16* bbase1 = cbh + (size_t)n1 * DD + g * 16;
#endif

        v8f acc0 = {};
        v8f acc1 = {};
#pragma unroll
        for (int c = 0; c < 8; ++c) {
            // B 32x16 layout: lane half g holds K = c*32 + g*16 + h (contiguous 16 halves)
            const _Float16* p0 = bbase0 + c * 32;
            const _Float16* p1 = bbase1 + c * 32;
            v8h b0lo = *(const v8h*)(p0);
            v8h b0hi = *(const v8h*)(p0 + 8);
            v8h b1lo = *(const v8h*)(p1);
            v8h b1hi = *(const v8h*)(p1 + 8);
            v16h bF0 = __builtin_shufflevector(b0lo, b0hi, 0,1,2,3,4,5,6,7,8,9,10,11,12,13,14,15);
            v16h bF1 = __builtin_shufflevector(b1lo, b1hi, 0,1,2,3,4,5,6,7,8,9,10,11,12,13,14,15);
            acc0 = __builtin_amdgcn_wmma_f32_16x16x32_f16(false, aF[c], false, bF0,
                                                          (short)0, acc0, false, false);
            acc1 = __builtin_amdgcn_wmma_f32_16x16x32_f16(false, aF[c], false, bF1,
                                                          (short)0, acc1, false, false);
        }

        const float cc0 = c2[n0];
        const float cc1 = c2[n1];
#pragma unroll
        for (int r = 0; r < 8; ++r) {
            // score = ||c||^2 - 2 x.c  (||x||^2 constant per row -> dropped)
            float s0 = cc0 - 2.f * acc0[r];
            float s1 = cc1 - 2.f * acc1[r];
            if (s0 < best[r]) { best[r] = s0; bidx[r] = n0; }
            if (s1 < best[r]) { best[r] = s1; bidx[r] = n1; }
        }

#if HAS_TDM
        __syncthreads();                 // everyone done reading buf before refill
        if (leader && s < 30) {
            tdm_load_cb_tile(cbh + (size_t)(s + 2) * 32 * DD, &ldsB[buf][0]);
        }
#endif
    }

    // min-reduce across the 16 lanes of each half (masks <=8 never cross halves)
#pragma unroll
    for (int r = 0; r < 8; ++r) {
#pragma unroll
        for (int m = 1; m < 16; m <<= 1) {
            float ob = __shfl_xor(best[r], m, 32);
            int   oi = __shfl_xor(bidx[r], m, 32);
            if (ob < best[r] || (ob == best[r] && oi < bidx[r])) {
                best[r] = ob; bidx[r] = oi;
            }
        }
    }

    if (nlo == 0) {
        // C/D layout: vgpr r -> M = r + 8*g
#pragma unroll
        for (int r = 0; r < 8; ++r) {
            int row = blk * 64 + wave * 16 + r + 8 * g;
            codesI[row] = bidx[r];
            codes_out_f[row] = (float)bidx[r];
        }
    }
}

// ---------------------------------------------------------------------------
// 3) fused gather + commitment loss + histogram + music projection/cosine
//    one thread per (b,t); single pass over x
// ---------------------------------------------------------------------------
__global__ __launch_bounds__(256) void maq_gather_music_kernel(const float* __restrict__ x,
                                                               const float* __restrict__ mctx,
                                                               const float* __restrict__ codebook,
                                                               const float* __restrict__ w_proj,
                                                               const float* __restrict__ b_proj,
                                                               char* __restrict__ ws,
                                                               float* __restrict__ outQ) {
    int* codesI   = (int*)(ws + WS_CODES);
    float* counts = (float*)(ws + WS_COUNTS);
    float* accum  = (float*)(ws + WS_ACCUM);
    float* msim   = (float*)(ws + WS_MSIM);

    __shared__ float wl[3 * DD];
    __shared__ float red[256];
    const int tid = threadIdx.x;
    const int b = blockIdx.x >> 3;
    const int t = (blockIdx.x & 7) * 256 + tid;

    for (int i = tid; i < 3 * DD; i += 256) wl[i] = w_proj[i];
    __syncthreads();

    const int code = codesI[b * TT + t];
    const float* cbrow = codebook + (size_t)code * DD;
    float csum = 0.f, p0 = 0.f, p1 = 0.f, p2 = 0.f;

    for (int d = 0; d < DD; ++d) {
        float xv = x[((size_t)b * DD + d) * TT + t];
        float q  = cbrow[d];
        outQ[((size_t)b * DD + d) * TT + t] = q;   // quantized_st == quantized numerically
        float df = q - xv;
        csum += df * df;
        p0 += xv * wl[d];
        p1 += xv * wl[DD + d];
        p2 += xv * wl[2 * DD + d];
    }
    p0 += b_proj[0]; p1 += b_proj[1]; p2 += b_proj[2];

    float m0 = mctx[((size_t)b * 3 + 0) * TT + t];
    float m1 = mctx[((size_t)b * 3 + 1) * TT + t];
    float m2 = mctx[((size_t)b * 3 + 2) * TT + t];
    float num = p0 * m0 + p1 * m1 + p2 * m2;
    float npn = fmaxf(sqrtf(p0 * p0 + p1 * p1 + p2 * p2), 1e-8f);
    float ncn = fmaxf(sqrtf(m0 * m0 + m1 * m1 + m2 * m2), 1e-8f);
    msim[b * TT + t] = num / (npn * ncn);

    atomicAdd(&counts[code], 1.0f);

    red[tid] = csum; __syncthreads();
    for (int off = 128; off > 0; off >>= 1) {
        if (tid < off) red[tid] += red[tid + off];
        __syncthreads();
    }
    if (tid == 0) atomicAdd(&accum[0], red[0]);
}

// ---------------------------------------------------------------------------
// 4) music consistency loss: sum |dSim| * (code changed)
// ---------------------------------------------------------------------------
__global__ __launch_bounds__(256) void maq_music_loss_kernel(char* __restrict__ ws) {
    const float* msim = (const float*)(ws + WS_MSIM);
    const int* codesI = (const int*)(ws + WS_CODES);
    float* accum      = (float*)(ws + WS_ACCUM);
    __shared__ float red[256];
    const int i = blockIdx.x * 256 + threadIdx.x;
    const int t = i & (TT - 1);
    float v = 0.f;
    if (t > 0) {
        float d = fabsf(msim[i] - msim[i - 1]);
        v = (codesI[i] != codesI[i - 1]) ? d : 0.f;
    }
    red[threadIdx.x] = v; __syncthreads();
    for (int off = 128; off > 0; off >>= 1) {
        if (threadIdx.x < off) red[threadIdx.x] += red[threadIdx.x + off];
        __syncthreads();
    }
    if (threadIdx.x == 0) atomicAdd(&accum[1], red[0]);
}

// ---------------------------------------------------------------------------
// 5) finalize: perplexity + scalar outputs
// ---------------------------------------------------------------------------
__global__ __launch_bounds__(256) void maq_finalize_kernel(char* __restrict__ ws,
                                                           float* __restrict__ out_tail) {
    const float* counts = (const float*)(ws + WS_COUNTS);
    const float* accum  = (const float*)(ws + WS_ACCUM);
    __shared__ float red[256];
    const int tid = threadIdx.x;
    float e = 0.f;
    for (int k = tid; k < KK; k += 256) {
        float p = counts[k] * (1.0f / (float)MM);
        e += p * logf(p + 1e-10f);
    }
    red[tid] = e; __syncthreads();
    for (int off = 128; off > 0; off >>= 1) {
        if (tid < off) red[tid] += red[tid + off];
        __syncthreads();
    }
    if (tid == 0) {
        float commit = accum[0] / ((float)MM * (float)DD);
        float mloss  = accum[1] / ((float)BB * (float)(TT - 1));
        out_tail[0] = commit + 0.1f * mloss;  // commitment_total
        out_tail[1] = commit;                 // codebook_loss (same value, sg only affects grads)
        out_tail[2] = expf(-red[0]);          // perplexity
        out_tail[3] = mloss;                  // music_loss
    }
}

// ---------------------------------------------------------------------------
extern "C" void kernel_launch(void* const* d_in, const int* in_sizes, int n_in,
                              void* d_out, int out_size, void* d_ws, size_t ws_size,
                              hipStream_t stream) {
    const float* x        = (const float*)d_in[0];  // [16,256,2048]
    const float* mctx     = (const float*)d_in[1];  // [16,3,2048]
    const float* codebook = (const float*)d_in[2];  // [1024,256]
    const float* w_proj   = (const float*)d_in[3];  // [3,256]
    const float* b_proj   = (const float*)d_in[4];  // [3]
    (void)in_sizes; (void)n_in; (void)out_size; (void)ws_size;

    float* outQ     = (float*)d_out;                       // quantized_st [16,256,2048]
    float* codes_f  = outQ + (size_t)BB * DD * TT;         // codes [16,2048] as float
    float* out_tail = codes_f + MM;                        // 4 scalars
    char* ws = (char*)d_ws;

    maq_prep_kernel<<<KK, 128, 0, stream>>>(codebook, ws);
    maq_gemm_argmin_kernel<<<MM / 64, 128, 0, stream>>>(x, ws, codes_f);
    maq_gather_music_kernel<<<BB * TT / 256, 256, 0, stream>>>(x, mctx, codebook,
                                                               w_proj, b_proj, ws, outQ);
    maq_music_loss_kernel<<<MM / 256, 256, 0, stream>>>(ws);
    maq_finalize_kernel<<<1, 256, 0, stream>>>(ws, out_tail);
}
